// MoELayerSharded_53154515256361
// MI455X (gfx1250) — matrix operand order repaired and checked by reading, not compiled, not run on hardware
//
#include <hip/hip_runtime.h>

// ---------------- problem constants (from reference) ----------------
#define DD   1024          // n_embd
#define HH   4096          // SwiGLU hidden
#define EE   8             // experts
#define TT   8192          // B*S tokens
#define MTILE 32           // tokens per workgroup tile
#define MAX_TILES 256      // worst case TT/MTILE tiles per expert

typedef __bf16 v16bf __attribute__((ext_vector_type(16)));
typedef __bf16 v8bf  __attribute__((ext_vector_type(8)));
typedef float  v8f   __attribute__((ext_vector_type(8)));

// ---------------- 0) zero output + counters ----------------
__global__ void moe_zero_kernel(float* __restrict__ y, int* __restrict__ counts) {
    int i = blockIdx.x * blockDim.x + threadIdx.x;          // 8192*256 = TT*DD/4
    if (blockIdx.x == 0 && threadIdx.x < EE) counts[threadIdx.x] = 0;
    ((float4*)y)[i] = make_float4(0.f, 0.f, 0.f, 0.f);
}

// ---------------- 1) x fp32 -> bf16 ----------------
__global__ void moe_cvt_kernel(const float* __restrict__ x, __bf16* __restrict__ xb) {
    int i = (blockIdx.x * blockDim.x + threadIdx.x) * 8;    // 4096*256*8 = TT*DD
    float4 a = *(const float4*)(x + i);
    float4 b = *(const float4*)(x + i + 4);
    v8bf v;
    v[0] = (__bf16)a.x; v[1] = (__bf16)a.y; v[2] = (__bf16)a.z; v[3] = (__bf16)a.w;
    v[4] = (__bf16)b.x; v[5] = (__bf16)b.y; v[6] = (__bf16)b.z; v[7] = (__bf16)b.w;
    *(v8bf*)(xb + i) = v;
}

// ---------------- 2) gate: logits, top-2, softmax, dispatch lists ----------------
__global__ void __launch_bounds__(256)
moe_gate_kernel(const float* __restrict__ x, const float* __restrict__ noise,
                const float* __restrict__ gate_w, const float* __restrict__ noise_w,
                int* __restrict__ counts, int* __restrict__ tlist, float* __restrict__ wlist) {
    __shared__ float gw[DD * EE];                           // 32 KB
    for (int i = threadIdx.x; i < DD * EE; i += 256) gw[i] = gate_w[i];
    __syncthreads();

    int token = blockIdx.x * 256 + threadIdx.x;             // grid = 32 -> exactly TT
    const float* xr = x + (size_t)token * DD;
    float acc[EE];
#pragma unroll
    for (int e = 0; e < EE; ++e) acc[e] = 0.f;
    for (int d = 0; d < DD; ++d) {
        float xv = xr[d];
#pragma unroll
        for (int e = 0; e < EE; ++e) acc[e] += xv * gw[d * EE + e];
    }
#pragma unroll
    for (int e = 0; e < EE; ++e) acc[e] += noise[(size_t)token * EE + e] * noise_w[e];

    // top-2, lowest index wins ties (matches lax.top_k)
    int i0 = 0;
#pragma unroll
    for (int e = 1; e < EE; ++e) if (acc[e] > acc[i0]) i0 = e;
    int i1 = (i0 == 0) ? 1 : 0;
#pragma unroll
    for (int e = 0; e < EE; ++e) if (e != i0 && acc[e] > acc[i1]) i1 = e;

    float e1   = __expf(acc[i1] - acc[i0]);
    float rs   = __builtin_amdgcn_rcpf(1.0f + e1);
    float w0   = rs;
    float w1v  = e1 * rs;

    int s0 = atomicAdd(&counts[i0], 1);
    tlist[i0 * TT + s0] = token;  wlist[i0 * TT + s0] = w0;
    int s1 = atomicAdd(&counts[i1], 1);
    tlist[i1 * TT + s1] = token;  wlist[i1 * TT + s1] = w1v;
}

// ---------------- 3) fused gathered SwiGLU + down-proj, bf16 WMMA ----------------
__global__ void __launch_bounds__(256)
moe_expert_kernel(const __bf16* __restrict__ xb,
                  const float* __restrict__ w1, const float* __restrict__ b1,
                  const float* __restrict__ w2, const float* __restrict__ b2,
                  const float* __restrict__ wp, const float* __restrict__ bp,
                  const int* __restrict__ counts, const int* __restrict__ tlist,
                  const float* __restrict__ wlist, float* __restrict__ y) {
    const int e    = blockIdx.x >> 8;                       // / MAX_TILES
    const int tile = blockIdx.x & (MAX_TILES - 1);
    const int nt   = counts[e];
    if (tile * MTILE >= nt) return;

    __shared__ __bf16 xs[MTILE * DD];                       // 64 KB gathered x tile
    __shared__ __bf16 hs[MTILE * 128];                      // 8 KB  h chunk
    __shared__ float  gws[MTILE];
    __shared__ int    toks[MTILE];

    const int tid = threadIdx.x;
    if (tid < MTILE) {
        int s   = tile * MTILE + tid;
        int idx = (s < nt) ? s : (nt - 1);
        toks[tid] = tlist[e * TT + idx];
        gws[tid]  = (s < nt) ? wlist[e * TT + idx] : 0.0f;
    }
    __syncthreads();

    {   // async gather: 32 token rows (2 KB each) of bf16 x -> LDS, no VGPR round-trip
        int row  = tid >> 3;                                // 0..31
        int part = tid & 7;                                 // 256 B per thread
        unsigned long long gsrc =
            (unsigned long long)(uintptr_t)(xb + (size_t)toks[row] * DD) + part * 256;
        unsigned ldst = (unsigned)(uintptr_t)(xs + row * DD) + part * 256;
#pragma unroll
        for (int i = 0; i < 16; ++i) {
            asm volatile("global_load_async_to_lds_b128 %0, %1, off"
                         :: "v"(ldst + i * 16), "v"(gsrc + (unsigned long long)i * 16)
                         : "memory");
        }
        asm volatile("s_wait_asynccnt 0" ::: "memory");
    }
    __syncthreads();

    const int wave  = tid >> 5;                             // 0..7
    const int lane  = tid & 31;
    const int lhalf = lane & 15;
    const bool hi   = lane >= 16;

    const float* w1e = w1 + (size_t)e * DD * HH;
    const float* w2e = w2 + (size_t)e * DD * HH;
    const float* b1e = b1 + (size_t)e * HH;
    const float* b2e = b2 + (size_t)e * HH;
    const float* wpe = wp + (size_t)e * HH * DD;
    const float* bpe = bp + (size_t)e * DD;

    v8f acc_y[8][2] = {};                                   // 32 x (wave's 128 cols) fp32

    for (int hc = 0; hc < HH; hc += 128) {                  // 32 chunks
        // ---- up-projection: this wave owns h columns [hc + wave*16, +16) ----
        v8f c1[2] = {}, c2[2] = {};
        const int ncol = hc + wave * 16 + lhalf;
        for (int kb = 0; kb < DD; kb += 32) {
            const int krb = kb + (hi ? 16 : 0);
            const float* p1 = w1e + (size_t)krb * HH + ncol;
            const float* p2 = w2e + (size_t)krb * HH + ncol;
            __builtin_prefetch(p1 + 32 * HH, 0, 1);         // next K-block (global_prefetch_b8)
            __builtin_prefetch(p2 + 32 * HH, 0, 1);
            v16bf B1, B2;
#pragma unroll
            for (int r = 0; r < 8; ++r) {
                B1[2*r]   = (__bf16)p1[(size_t)(2*r)     * HH];
                B1[2*r+1] = (__bf16)p1[(size_t)(2*r + 1) * HH];
                B2[2*r]   = (__bf16)p2[(size_t)(2*r)     * HH];
                B2[2*r+1] = (__bf16)p2[(size_t)(2*r + 1) * HH];
            }
            const int k0 = kb + (hi ? 16 : 0);
            const int k1 = kb + (hi ? 24 : 8);
#pragma unroll
            for (int mt = 0; mt < 2; ++mt) {                // B reused across 2 M-tiles
                const int m = mt * 16 + lhalf;
                v8bf alo = *(const v8bf*)(xs + m * DD + k0);
                v8bf ahi = *(const v8bf*)(xs + m * DD + k1);
                v16bf A = __builtin_shufflevector(alo, ahi,
                          0,1,2,3,4,5,6,7,8,9,10,11,12,13,14,15);
                c1[mt] = __builtin_amdgcn_wmma_f32_16x16x32_bf16(false, A, false, B1, (short)0, c1[mt], false, false);
                c2[mt] = __builtin_amdgcn_wmma_f32_16x16x32_bf16(false, A, false, B2, (short)0, c2[mt], false, false);
            }
        }
        // bias + SwiGLU -> bf16 h chunk in LDS (single v_rcp, no IEEE divide)
        const float bb1 = b1e[ncol];
        const float bb2 = b2e[ncol];
#pragma unroll
        for (int mt = 0; mt < 2; ++mt) {
#pragma unroll
            for (int r = 0; r < 8; ++r) {
                float u = c1[mt][r] + bb1;
                float v = c2[mt][r] + bb2;
                float h = u * v * __builtin_amdgcn_rcpf(1.0f + __expf(-v));
                hs[(mt * 16 + r + (hi ? 8 : 0)) * 128 + wave * 16 + lhalf] = (__bf16)h;
            }
        }
        __syncthreads();

        // ---- down-projection: this wave owns D columns [wave*128, +128) ----
#pragma unroll
        for (int ntile = 0; ntile < 8; ++ntile) {
            const int col = wave * 128 + ntile * 16 + lhalf;
#pragma unroll
            for (int ks = 0; ks < 128; ks += 32) {
                const int krb = hc + ks + (hi ? 16 : 0);
                const float* p = wpe + (size_t)krb * DD + col;
                __builtin_prefetch(p + 32 * DD, 0, 1);
                v16bf B;
#pragma unroll
                for (int r = 0; r < 8; ++r) {
                    B[2*r]   = (__bf16)p[(size_t)(2*r)     * DD];
                    B[2*r+1] = (__bf16)p[(size_t)(2*r + 1) * DD];
                }
                const int k0 = ks + (hi ? 16 : 0);
                const int k1 = ks + (hi ? 24 : 8);
#pragma unroll
                for (int mt = 0; mt < 2; ++mt) {            // B reused across 2 M-tiles
                    const int m = mt * 16 + lhalf;
                    v8bf alo = *(const v8bf*)(hs + m * 128 + k0);
                    v8bf ahi = *(const v8bf*)(hs + m * 128 + k1);
                    v16bf A = __builtin_shufflevector(alo, ahi,
                              0,1,2,3,4,5,6,7,8,9,10,11,12,13,14,15);
                    acc_y[ntile][mt] = __builtin_amdgcn_wmma_f32_16x16x32_bf16(false, A, false, B, (short)0, acc_y[ntile][mt], false, false);
                }
            }
        }
        __syncthreads();                                    // hs reused next chunk
    }

    // ---- epilogue: + bp, * gate weight, scatter-add into y ----
#pragma unroll
    for (int ntile = 0; ntile < 8; ++ntile) {
        const int col = wave * 128 + ntile * 16 + lhalf;
        const float bpv = bpe[col];
#pragma unroll
        for (int mt = 0; mt < 2; ++mt) {
#pragma unroll
            for (int r = 0; r < 8; ++r) {
                const int m = mt * 16 + r + (hi ? 8 : 0);
                float val = gws[m] * (acc_y[ntile][mt][r] + bpv);
                atomicAdd(&y[(size_t)toks[m] * DD + col], val);
            }
        }
    }
}

// ---------------- launch ----------------
extern "C" void kernel_launch(void* const* d_in, const int* in_sizes, int n_in,
                              void* d_out, int out_size, void* d_ws, size_t ws_size,
                              hipStream_t stream) {
    const float* x       = (const float*)d_in[0];
    const float* noise   = (const float*)d_in[1];
    const float* gate_w  = (const float*)d_in[2];
    const float* noise_w = (const float*)d_in[3];
    const float* w1      = (const float*)d_in[4];
    const float* b1      = (const float*)d_in[5];
    const float* w2      = (const float*)d_in[6];
    const float* b2      = (const float*)d_in[7];
    const float* wp      = (const float*)d_in[8];
    const float* bp      = (const float*)d_in[9];
    float* y = (float*)d_out;

    // workspace layout (~17.3 MB)
    char* ws = (char*)d_ws;
    const size_t XB_BYTES = (size_t)TT * DD * 2;            // 16 MB bf16 x
    const size_t TL_BYTES = (size_t)EE * TT * 4;            // 256 KB token lists
    const size_t WL_BYTES = (size_t)EE * TT * 4;            // 256 KB gate weights
    __bf16* xb    = (__bf16*)ws;
    int*    tlist = (int*)(ws + XB_BYTES);
    float*  wlist = (float*)(ws + XB_BYTES + TL_BYTES);
    int*    counts = (int*)(ws + XB_BYTES + TL_BYTES + WL_BYTES);

    moe_zero_kernel<<<(TT * DD / 4) / 256, 256, 0, stream>>>(y, counts);
    moe_cvt_kernel<<<(TT * DD / 8) / 256, 256, 0, stream>>>(x, xb);
    moe_gate_kernel<<<TT / 256, 256, 0, stream>>>(x, noise, gate_w, noise_w, counts, tlist, wlist);
    moe_expert_kernel<<<EE * MAX_TILES, 256, 0, stream>>>(xb, w1, b1, w2, b2, wp, bp,
                                                          counts, tlist, wlist, y);
}
